// RPN_43568148251506
// MI455X (gfx1250) — matrix operand order repaired and checked by reading, not compiled, not run on hardware
//
#include <hip/hip_runtime.h>
#include <hip/hip_bf16.h>
#include <math.h>

// ---------------------------------------------------------------------------
// RPN forward for MI455X (gfx1250, wave32, WMMA).
//  conv3x3  : implicit GEMM, bf16 WMMA fp32-accum, 128x64 block tile,
//             LDS pre-swizzled into per-lane fragment order (ds_load_b128).
//  heads    : fused 1x1 cls+bbox GEMM (M=54 pad 64), same technique.
//  decode   : 2-class softmax + anchor decode.
//  sort     : bitonic, (score desc, idx asc) == stable argsort(-scores).
//  nms      : parallel IoU suppression bitmatrix + single-block greedy scan.
// ---------------------------------------------------------------------------

typedef __attribute__((ext_vector_type(16))) __bf16 v16bf;
typedef __attribute__((ext_vector_type(8)))  float  v8f;

#define NP    1024           // 32x32 positions
#define CIN   2048
#define NA    9
#define NBOX  (NP * NA)      // 9216
#define NSORT 16384          // next pow2 >= NBOX
#define KEEPW (NBOX / 32)    // 288 words (exact)
#define IOU_THR 0.5f

// pack two fp32 -> two bf16 (RNE) in one u32 (lo = even element)
static __device__ __forceinline__ unsigned pack2bf(float lo, float hi) {
  union { float f; unsigned u; } a, b; a.f = lo; b.f = hi;
  unsigned ra = a.u + 0x7FFFu + ((a.u >> 16) & 1u);
  unsigned rb = b.u + 0x7FFFu + ((b.u >> 16) & 1u);
  return (ra >> 16) | (rb & 0xFFFF0000u);
}

// ---------------------------------------------------------------------------
// Kernel 1: 3x3 conv as implicit GEMM.  M=512(Cout) N=1024(pos) K=2048*9.
// Block tile 128x64, 8 waves; each wave computes 32x32 (4 WMMA accumulators).
// LDS tiles are stored pre-swizzled in WMMA fragment order:
//   sA[mtile][lane][8 dwords] : A 16x32 bf16 fragment for that lane
//   sB[ntile][lane][8 dwords] : B 32x16 bf16 fragment for that lane
// A layout: lane<16 -> row=lane,   K = {0..7,16..23}  (elem e: K=(e&7)+16*(e>>3))
//           lane>=16 -> row=lane-16, K = {8..15,24..31}
// B layout: lane<16 -> N=lane, K = {0..7,16..23}; lane>=16 -> K = {8..15,24..31} (same as A)
// ---------------------------------------------------------------------------
__global__ __launch_bounds__(256)
void conv3x3_wmma(const float* __restrict__ x, const float* __restrict__ w,
                  const float* __restrict__ bias, float* __restrict__ feat) {
  __shared__ __align__(32) unsigned sA[8][32][8];  // 8 M-tiles x lane x 8 dw
  __shared__ __align__(32) unsigned sB[4][32][8];  // 4 N-tiles x lane x 8 dw
  const int tid  = threadIdx.x;
  const int lane = tid & 31;
  const int wv   = tid >> 5;
  const int co0  = blockIdx.x * 128;
  const int n0   = blockIdx.y * 64;
  const int ma   = (wv & 3) * 2;       // wave's first M tile (of 8)
  const int na   = (wv >> 2) * 2;      // wave's first N tile (of 4)
  v8f acc00 = {}, acc01 = {}, acc10 = {}, acc11 = {};

  for (int kh = 0; kh < 3; ++kh)
  for (int kw = 0; kw < 3; ++kw)
  for (int c0 = 0; c0 < CIN; c0 += 32) {
    // --- stage A: weights [128 co][32 ci], bf16-pair packed, swizzled ---
    #pragma unroll
    for (int r = 0; r < 8; ++r) {
      int pi = r * 256 + tid;            // 2048 pairs
      int i  = pi >> 4;                  // 0..127 (co within tile)
      int k  = (pi & 15) * 2;            // 0..30 even (ci within chunk)
      size_t base = (size_t)(co0 + i) * (CIN * 9) +
                    (size_t)(c0 + k) * 9 + kh * 3 + kw;
      float v0 = w[base];
      float v1 = w[base + 9];
      int half   = (k >> 3) & 1;                       // K in {8..15,24..31}?
      int lane_d = (i & 15) + 16 * half;
      int ew     = ((k & 7) + ((k >> 4) << 3)) >> 1;   // dword slot
      sA[i >> 4][lane_d][ew] = pack2bf(v0, v1);
    }
    if (c0 + 32 < CIN)   // global_prefetch_b8 of next weight chunk
      __builtin_prefetch(&w[(size_t)(co0 + (tid >> 1)) * (CIN * 9) +
                            (size_t)(c0 + 32 + (tid & 1) * 16) * 9 + kh * 3 + kw], 0, 1);
    // --- stage B: shifted activations [32 ci][64 pos], SAME padding ---
    #pragma unroll
    for (int r = 0; r < 4; ++r) {
      int pi = r * 256 + tid;            // 1024 pairs
      int j  = pi & 63;                  // position within tile
      int k  = (pi >> 6) * 2;            // 0..30 even (ci)
      int p  = n0 + j;
      int y  = (p >> 5) + kh - 1;
      int xx = (p & 31) + kw - 1;
      float v0 = 0.0f, v1 = 0.0f;
      if ((unsigned)y < 32u && (unsigned)xx < 32u) {
        size_t bi = ((size_t)(c0 + k) * 32 + y) * 32 + xx;
        v0 = x[bi];
        v1 = x[bi + 1024];               // next input channel
      }
      int lane_d = (j & 15) + 16 * ((k >> 3) & 1);
      int ew     = ((k & 7) + ((k >> 4) << 3)) >> 1;
      sB[j >> 4][lane_d][ew] = pack2bf(v0, v1);
    }
    __syncthreads();

    // --- contiguous fragment loads (2x ds_load_b128 each) + 4 WMMA ---
    v16bf a0 = *(const v16bf*)&sA[ma][lane][0];
    v16bf a1 = *(const v16bf*)&sA[ma + 1][lane][0];
    v16bf b0 = *(const v16bf*)&sB[na][lane][0];
    v16bf b1 = *(const v16bf*)&sB[na + 1][lane][0];
    acc00 = __builtin_amdgcn_wmma_f32_16x16x32_bf16(false, a0, false, b0,
                                                    (short)0, acc00, false, false);
    acc01 = __builtin_amdgcn_wmma_f32_16x16x32_bf16(false, a0, false, b1,
                                                    (short)0, acc01, false, false);
    acc10 = __builtin_amdgcn_wmma_f32_16x16x32_bf16(false, a1, false, b0,
                                                    (short)0, acc10, false, false);
    acc11 = __builtin_amdgcn_wmma_f32_16x16x32_bf16(false, a1, false, b1,
                                                    (short)0, acc11, false, false);
    __syncthreads();
  }

  // epilogue: bias + relu; C/D layout M = vgpr + 8*(lane>=16), N = lane&15
  #pragma unroll
  for (int i = 0; i < 8; ++i) {
    int mrow = (lane < 16) ? i : (8 + i);
    int ncol = lane & 15;
    int m0 = co0 + ma * 16 + mrow;
    int m1 = m0 + 16;
    int nc0 = n0 + na * 16 + ncol;
    int nc1 = nc0 + 16;
    float b0v = bias[m0], b1v = bias[m1];
    float v;
    v = acc00[i] + b0v; feat[(size_t)m0 * NP + nc0] = v > 0.0f ? v : 0.0f;
    v = acc01[i] + b0v; feat[(size_t)m0 * NP + nc1] = v > 0.0f ? v : 0.0f;
    v = acc10[i] + b1v; feat[(size_t)m1 * NP + nc0] = v > 0.0f ? v : 0.0f;
    v = acc11[i] + b1v; feat[(size_t)m1 * NP + nc1] = v > 0.0f ? v : 0.0f;
  }
}

// ---------------------------------------------------------------------------
// Kernel 2: fused 1x1 heads. rows 0..17 = cls, 18..53 = bbox, pad to 64.
// M=64 N=1024 K=512.  Block 64x64, wave = 1 A frag x 2 B frags.
// ---------------------------------------------------------------------------
__global__ __launch_bounds__(256)
void heads_wmma(const float* __restrict__ feat,
                const float* __restrict__ cls_w, const float* __restrict__ cls_b,
                const float* __restrict__ bbox_w, const float* __restrict__ bbox_b,
                float* __restrict__ od) {
  __shared__ __align__(32) unsigned sA[4][32][8];
  __shared__ __align__(32) unsigned sB[4][32][8];
  const int tid  = threadIdx.x;
  const int lane = tid & 31;
  const int wv   = tid >> 5;
  const int n0   = blockIdx.x * 64;
  const int ma   = wv & 3;
  const int na   = (wv >> 2) * 2;
  v8f acc0 = {}, acc1 = {};

  for (int c0 = 0; c0 < 512; c0 += 32) {
    #pragma unroll
    for (int r = 0; r < 4; ++r) {
      int pi = r * 256 + tid;            // 1024 pairs
      int i  = pi >> 4;                  // 0..63
      int k  = (pi & 15) * 2;
      float v0 = 0.0f, v1 = 0.0f;
      if (i < 18) {
        v0 = cls_w[(size_t)i * 512 + c0 + k];
        v1 = cls_w[(size_t)i * 512 + c0 + k + 1];
      } else if (i < 54) {
        v0 = bbox_w[(size_t)(i - 18) * 512 + c0 + k];
        v1 = bbox_w[(size_t)(i - 18) * 512 + c0 + k + 1];
      }
      int half   = (k >> 3) & 1;
      int lane_d = (i & 15) + 16 * half;
      int ew     = ((k & 7) + ((k >> 4) << 3)) >> 1;
      sA[i >> 4][lane_d][ew] = pack2bf(v0, v1);
    }
    #pragma unroll
    for (int r = 0; r < 4; ++r) {
      int pi = r * 256 + tid;
      int j  = pi & 63;
      int k  = (pi >> 6) * 2;
      size_t bi = (size_t)(c0 + k) * NP + n0 + j;
      float v0 = feat[bi];
      float v1 = feat[bi + NP];
      int lane_d = (j & 15) + 16 * ((k >> 3) & 1);
      int ew     = ((k & 7) + ((k >> 4) << 3)) >> 1;
      sB[j >> 4][lane_d][ew] = pack2bf(v0, v1);
    }
    __syncthreads();
    v16bf af = *(const v16bf*)&sA[ma][lane][0];
    v16bf b0 = *(const v16bf*)&sB[na][lane][0];
    v16bf b1 = *(const v16bf*)&sB[na + 1][lane][0];
    acc0 = __builtin_amdgcn_wmma_f32_16x16x32_bf16(false, af, false, b0,
                                                   (short)0, acc0, false, false);
    acc1 = __builtin_amdgcn_wmma_f32_16x16x32_bf16(false, af, false, b1,
                                                   (short)0, acc1, false, false);
    __syncthreads();
  }
  #pragma unroll
  for (int i = 0; i < 8; ++i) {
    int m  = ma * 16 + ((lane < 16) ? i : (8 + i));
    int nc = n0 + na * 16 + (lane & 15);
    if (m < 54) {
      float b = (m < 18) ? cls_b[m] : bbox_b[m - 18];
      od[(size_t)m * NP + nc]      = acc0[i] + b;
      od[(size_t)m * NP + nc + 16] = acc1[i] + b;
    }
  }
}

// ---------------------------------------------------------------------------
// Kernel 3: 2-class softmax + anchor decode (anchors have no grid shift).
// ---------------------------------------------------------------------------
__global__ void decode_kernel(const float* __restrict__ od,
                              float* __restrict__ scores, float* __restrict__ boxes) {
  int idx = blockIdx.x * blockDim.x + threadIdx.x;
  if (idx >= NBOX) return;
  int p = idx / NA, a = idx % NA;
  const float scl[3] = {8.0f, 16.0f, 32.0f};
  const float rat[3] = {0.5f, 1.0f, 2.0f};
  float s = scl[a / 3], r = rat[a % 3];
  float sq = sqrtf(r);
  float aw = s * sq, ah = s / sq;     // anchor width/height; center = (0,0)
  float o0 = od[(size_t)a * NP + p];
  float o1 = od[(size_t)(9 + a) * NP + p];
  scores[idx] = 1.0f / (1.0f + expf(o0 - o1));
  float d0 = od[(size_t)(18 + a * 4 + 0) * NP + p];
  float d1 = od[(size_t)(18 + a * 4 + 1) * NP + p];
  float d2 = od[(size_t)(18 + a * 4 + 2) * NP + p];
  float d3 = od[(size_t)(18 + a * 4 + 3) * NP + p];
  float px = d0 * aw, py = d1 * ah;
  float pw = expf(d2) * aw, ph = expf(d3) * ah;
  boxes[idx * 4 + 0] = px - 0.5f * pw;
  boxes[idx * 4 + 1] = py - 0.5f * ph;
  boxes[idx * 4 + 2] = px + 0.5f * pw;
  boxes[idx * 4 + 3] = py + 0.5f * ph;
}

// ---------------------------------------------------------------------------
// Sort: bitonic over 16384 padded slots. Comparator = (score desc, idx asc).
// ---------------------------------------------------------------------------
__global__ void sort_init(const float* __restrict__ scores,
                          float* __restrict__ key, int* __restrict__ idx) {
  int i = blockIdx.x * blockDim.x + threadIdx.x;
  if (i < NSORT) {
    key[i] = (i < NBOX) ? scores[i] : -INFINITY;
    idx[i] = i;
  }
}

__global__ void bitonic_step(float* __restrict__ key, int* __restrict__ idx,
                             int j, int k) {
  int i = blockIdx.x * blockDim.x + threadIdx.x;
  int ixj = i ^ j;
  if (ixj > i && ixj < NSORT) {
    float ka = key[i], kb = key[ixj];
    int   ia = idx[i], ib = idx[ixj];
    bool aFirst = (ka > kb) || (ka == kb && ia < ib);
    bool up = ((i & k) == 0);
    if (up ? !aFirst : aFirst) {
      key[i] = kb; key[ixj] = ka;
      idx[i] = ib; idx[ixj] = ia;
    }
  }
}

__global__ void gather_sorted(const float* __restrict__ key, const int* __restrict__ idx,
                              const float* __restrict__ boxes,
                              float* __restrict__ bs, float* __restrict__ ss) {
  int i = blockIdx.x * blockDim.x + threadIdx.x;
  if (i >= NBOX) return;
  int s = idx[i];
  bs[i * 4 + 0] = boxes[s * 4 + 0];
  bs[i * 4 + 1] = boxes[s * 4 + 1];
  bs[i * 4 + 2] = boxes[s * 4 + 2];
  bs[i * 4 + 3] = boxes[s * 4 + 3];
  ss[i] = key[i];
}

// ---------------------------------------------------------------------------
// NMS: suppression bitmatrix (fully parallel) + single-block greedy scan.
// ---------------------------------------------------------------------------
__global__ void iou_sup(const float* __restrict__ bs, unsigned* __restrict__ sup) {
  int gid = blockIdx.x * blockDim.x + threadIdx.x;
  if (gid >= NBOX * KEEPW) return;
  int i = gid / KEEPW, w = gid % KEEPW;
  float x1 = bs[i * 4 + 0], y1 = bs[i * 4 + 1];
  float x2 = bs[i * 4 + 2], y2 = bs[i * 4 + 3];
  float ai = (x2 - x1) * (y2 - y1);
  unsigned bits = 0u;
  int jb = w * 32;
  #pragma unroll 4
  for (int b = 0; b < 32; ++b) {
    int j = jb + b;
    if (j <= i) continue;
    float u1 = bs[j * 4 + 0], v1 = bs[j * 4 + 1];
    float u2 = bs[j * 4 + 2], v2 = bs[j * 4 + 3];
    float aj = (u2 - u1) * (v2 - v1);
    float lx = fmaxf(x1, u1), ly = fmaxf(y1, v1);
    float rx = fminf(x2, u2), ry = fminf(y2, v2);
    float iw = fmaxf(rx - lx, 0.0f), ih = fmaxf(ry - ly, 0.0f);
    float inter = iw * ih;
    float uni = fmaxf(ai + aj - inter, 1e-9f);
    if (inter / uni > IOU_THR) bits |= (1u << b);
  }
  sup[gid] = bits;
}

__global__ __launch_bounds__(320)
void nms_greedy(const unsigned* __restrict__ sup, unsigned* __restrict__ keepout) {
  __shared__ unsigned keep[KEEPW];
  int t = threadIdx.x;
  if (t < KEEPW) keep[t] = 0xFFFFFFFFu;
  __syncthreads();
  for (int i = 0; i < NBOX; ++i) {
    unsigned ki = keep[i >> 5] & (1u << (i & 31));
    __syncthreads();
    if (ki && t < KEEPW) keep[t] &= ~sup[(size_t)i * KEEPW + t];
    __syncthreads();
  }
  if (t < KEEPW) keepout[t] = keep[t];
}

__global__ void finalize(const unsigned* __restrict__ keep,
                         const float* __restrict__ bs, const float* __restrict__ ss,
                         float* __restrict__ out) {
  int i = blockIdx.x * blockDim.x + threadIdx.x;
  if (i >= NBOX) return;
  bool k = (keep[i >> 5] >> (i & 31)) & 1u;
  out[i * 4 + 0] = k ? bs[i * 4 + 0] : 0.0f;
  out[i * 4 + 1] = k ? bs[i * 4 + 1] : 0.0f;
  out[i * 4 + 2] = k ? bs[i * 4 + 2] : 0.0f;
  out[i * 4 + 3] = k ? bs[i * 4 + 3] : 0.0f;
  out[NBOX * 4 + i] = k ? ss[i] : 0.0f;
}

// ---------------------------------------------------------------------------
extern "C" void kernel_launch(void* const* d_in, const int* in_sizes, int n_in,
                              void* d_out, int out_size, void* d_ws, size_t ws_size,
                              hipStream_t stream) {
  const float* x      = (const float*)d_in[0];
  const float* conv_w = (const float*)d_in[1];
  const float* conv_b = (const float*)d_in[2];
  const float* cls_w  = (const float*)d_in[3];
  const float* cls_b  = (const float*)d_in[4];
  const float* bbox_w = (const float*)d_in[5];
  const float* bbox_b = (const float*)d_in[6];

  char* ws = (char*)d_ws;
  size_t off = 0;
  auto alloc = [&](size_t bytes) -> void* {
    void* p = ws + off;
    off = (off + bytes + 255) & ~(size_t)255;
    return p;
  };
  float*    feat   = (float*)alloc((size_t)512 * NP * 4);
  float*    od     = (float*)alloc((size_t)64 * NP * 4);
  float*    scores = (float*)alloc((size_t)NBOX * 4);
  float*    boxes  = (float*)alloc((size_t)NBOX * 4 * 4);
  float*    skey   = (float*)alloc((size_t)NSORT * 4);
  int*      sidx   = (int*)alloc((size_t)NSORT * 4);
  float*    bs     = (float*)alloc((size_t)NBOX * 4 * 4);
  float*    ss     = (float*)alloc((size_t)NBOX * 4);
  unsigned* sup    = (unsigned*)alloc((size_t)NBOX * KEEPW * 4);
  unsigned* keepb  = (unsigned*)alloc((size_t)KEEPW * 4);

  conv3x3_wmma<<<dim3(4, 16), 256, 0, stream>>>(x, conv_w, conv_b, feat);
  heads_wmma<<<dim3(16), 256, 0, stream>>>(feat, cls_w, cls_b, bbox_w, bbox_b, od);
  decode_kernel<<<(NBOX + 255) / 256, 256, 0, stream>>>(od, scores, boxes);
  sort_init<<<NSORT / 256, 256, 0, stream>>>(scores, skey, sidx);
  for (int k = 2; k <= NSORT; k <<= 1)
    for (int j = k >> 1; j > 0; j >>= 1)
      bitonic_step<<<NSORT / 256, 256, 0, stream>>>(skey, sidx, j, k);
  gather_sorted<<<(NBOX + 255) / 256, 256, 0, stream>>>(skey, sidx, boxes, bs, ss);
  iou_sup<<<((NBOX * KEEPW) + 255) / 256, 256, 0, stream>>>(bs, sup);
  nms_greedy<<<1, 320, 0, stream>>>(sup, keepb);
  finalize<<<(NBOX + 255) / 256, 256, 0, stream>>>(keepb, bs, ss, (float*)d_out);
}
